// WindowAttention_69449621176982
// MI455X (gfx1250) — compile-verified, matrix operand order checked by reading, not run on hardware
//
#include <hip/hip_runtime.h>

typedef __attribute__((ext_vector_type(16))) __bf16 bf16x16;
typedef __attribute__((ext_vector_type(8)))  __bf16 bf16x8;
typedef __attribute__((ext_vector_type(8)))  float  f32x8;
typedef int v4i_gcc __attribute__((__vector_size__(16)));   // matches builtin proto

#define WIN_N 49
#define NP    64
#define DIMC  512
#define NH    16
#define HD    32
#define SCALE 0.17677669529663687f   // 32^-0.5

#define GLOBAL_AS __attribute__((address_space(1)))
#define LDS_AS    __attribute__((address_space(3)))

#if __has_builtin(__builtin_amdgcn_global_load_async_to_lds_b128) && \
    __has_builtin(__builtin_amdgcn_s_wait_asynccnt)
#define HAVE_ASYNC_LDS 1
#else
#define HAVE_ASYNC_LDS 0
#endif

// Native RNE conversion: LLVM fptrunc f32->bf16 lowers to v_cvt_pk_bf16_f32.
__device__ inline __bf16 to_bf16(float f) { return (__bf16)f; }

union FragU { bf16x16 v; bf16x8 h[2]; };

// A/B fragment loader for v_wmma_f32_16x16x32_bf16.
// Lane l (wave32): row/col = l&15, K-chunks at kbase=(l>>4)*8 and kbase+16.
__device__ inline bf16x16 load_frag(const __bf16* p, int row, int stride, int kbase) {
    FragU f;
    const __bf16* q = p + row * stride + kbase;
    f.h[0] = *reinterpret_cast<const bf16x8*>(q);
    f.h[1] = *reinterpret_cast<const bf16x8*>(q + 16);
    return f.v;
}

__device__ inline f32x8 wmma_bf16(bf16x16 a, bf16x16 b, f32x8 c) {
    // (neg_a, A, neg_b, B, c_mod, C, reuse_a, reuse_b)
    return __builtin_amdgcn_wmma_f32_16x16x32_bf16(false, a, false, b, (short)0, c, false, false);
}

// ---------------------------------------------------------------------------
// Kernel 1: fused per-window  qkv-projection + rel-pos-bias + mask + softmax
//           + P@V.  One block per window (4096 blocks, 256 threads = 8 waves).
// Writes attention output (B_*N, C) as bf16 into workspace.
// ---------------------------------------------------------------------------
__global__ __launch_bounds__(256) void swin_qkv_attn_kernel(
    const float* __restrict__ x, const float* __restrict__ mask,
    const float* __restrict__ qkv_w, const float* __restrict__ qkv_b,
    const float* __restrict__ bias_table, __bf16* __restrict__ attn_out)
{
    __shared__ __align__(16) __bf16 Xt[NP][32];   // x K-chunk tile (M x K)
    __shared__ __align__(16) __bf16 Wt[96][32];   // W^T tile (q|k|v cols for head)
    __shared__ __align__(16) __bf16 Qs[NP][32];   // q (scaled)      [token][hd]
    __shared__ __align__(16) __bf16 Ks[NP][32];   // k               [token][hd]
    __shared__ __align__(16) __bf16 Vt[32][NP];   // v transposed    [hd][token]
    __shared__ float  Sf[NP][NP];                 // attn scores (fp32)
    __shared__ __align__(16) __bf16 Ps[NP][NP];   // softmax probs

    const int w     = blockIdx.x;
    const int tid   = threadIdx.x;
    const int lane  = tid & 31;
    const int wid   = tid >> 5;
    const int lrow  = lane & 15;
    const int kb    = (lane >> 4) * 8;
    const int rbase = (lane >> 4) * 8;            // C-frag row base
    const int mt    = wid & 3;
    const int nhalf = wid >> 2;                   // 0 or 1 (wave-uniform)

    for (int h = 0; h < NH; ++h) {
        f32x8 acc0 = {}, acc1 = {}, acc2 = {};
        // ---- qkv projection for this head: (64x512) @ (512x96) ----
        for (int k0 = 0; k0 < DIMC; k0 += 32) {
            { // stage X tile, fp32 -> bf16, pad rows >= 49 with zero
                int row = tid >> 2;
                int cg  = (tid & 3) * 8;
                float v0[8];
                if (row < WIN_N) {
                    const float4* src = reinterpret_cast<const float4*>(
                        x + (size_t)(w * WIN_N + row) * DIMC + k0 + cg);
                    float4 f0 = src[0], f1 = src[1];
                    v0[0]=f0.x; v0[1]=f0.y; v0[2]=f0.z; v0[3]=f0.w;
                    v0[4]=f1.x; v0[5]=f1.y; v0[6]=f1.z; v0[7]=f1.w;
                } else {
                    for (int i = 0; i < 8; ++i) v0[i] = 0.f;
                }
                for (int i = 0; i < 8; ++i) Xt[row][cg + i] = to_bf16(v0[i]);
            }
            // stage W^T tile: Wt[n][kk].  Load float4 along n (contiguous in
            // qkv_w rows; a 4-aligned quad never straddles a q/k/v section),
            // scatter 4 bf16 into the transposed LDS tile.
            for (int i = 0; i < 3; ++i) {
                int e   = tid + i * 256;     // 0..767
                int kk  = e & 31;            // K row
                int nq4 = (e >> 5) * 4;      // n quad base: 0,4,...,92
                int c   = (nq4 >> 5) * DIMC + h * HD + (nq4 & 31);
                const float4* src = reinterpret_cast<const float4*>(
                    qkv_w + (size_t)(k0 + kk) * (3 * DIMC) + c);
                float4 f = *src;
                Wt[nq4 + 0][kk] = to_bf16(f.x);
                Wt[nq4 + 1][kk] = to_bf16(f.y);
                Wt[nq4 + 2][kk] = to_bf16(f.z);
                Wt[nq4 + 3][kk] = to_bf16(f.w);
            }
            // prefetch next K-step of the weight stream (L2-resident hot path)
            if (k0 + 32 < DIMC) {
                __builtin_prefetch(qkv_w + (size_t)(k0 + 32 + (tid >> 3)) * (3 * DIMC)
                                         + (tid & 7) * 4, 0, 1);
            }
            __syncthreads();
            int nb = nhalf * 48;
            bf16x16 a  = load_frag(&Xt[0][0], mt * 16 + lrow, 32, kb);
            bf16x16 b0 = load_frag(&Wt[0][0], nb + lrow,      32, kb);
            bf16x16 b1 = load_frag(&Wt[0][0], nb + 16 + lrow, 32, kb);
            bf16x16 b2 = load_frag(&Wt[0][0], nb + 32 + lrow, 32, kb);
            acc0 = wmma_bf16(a, b0, acc0);
            acc1 = wmma_bf16(a, b1, acc1);
            acc2 = wmma_bf16(a, b2, acc2);
            __syncthreads();
        }
        // ---- scatter q,k,v to LDS (+qkv bias, q scaled, v transposed) ----
        // Section and column-base are WAVE-UNIFORM (scalar branches, no EXEC
        // divergence): cbase = nhalf*48 + j*16 never straddles a 32-col section.
        {
            for (int j = 0; j < 3; ++j) {
                f32x8 accj = (j == 0) ? acc0 : (j == 1) ? acc1 : acc2;
                int cbase = nhalf * 48 + j * 16;   // multiple of 16
                int sec   = cbase >> 5;            // uniform: 0=q 1=k 2=v
                int cc    = (cbase & 31) + lrow;   // lane column within section
                float bia = qkv_b[sec * DIMC + h * HD + cc];
                if (sec == 0) {
                    for (int r = 0; r < 8; ++r)
                        Qs[mt * 16 + rbase + r][cc] = to_bf16((accj[r] + bia) * SCALE);
                } else if (sec == 1) {
                    for (int r = 0; r < 8; ++r)
                        Ks[mt * 16 + rbase + r][cc] = to_bf16(accj[r] + bia);
                } else {
                    for (int r = 0; r < 8; ++r)
                        Vt[cc][mt * 16 + rbase + r] = to_bf16(accj[r] + bia);
                }
            }
        }
        __syncthreads();
        // ---- S = Q @ K^T (64x64), add rel-pos bias + shift mask ----
        {
            bf16x16 a = load_frag(&Qs[0][0], mt * 16 + lrow, 32, kb);
            for (int jn = 0; jn < 2; ++jn) {
                int nt = nhalf * 2 + jn;
                bf16x16 b = load_frag(&Ks[0][0], nt * 16 + lrow, 32, kb);
                f32x8 c = {};
                c = wmma_bf16(a, b, c);
                for (int r = 0; r < 8; ++r) {
                    int row = mt * 16 + rbase + r;
                    int col = nt * 16 + lrow;
                    float s = -1e30f;
                    if (row < WIN_N && col < WIN_N) {
                        int ri = row / 7, ci = row % 7;
                        int rj = col / 7, cj = col % 7;
                        int idx = (ci - cj + 6) * 13 + (ri - rj + 6);
                        s = c[r] + bias_table[idx * NH + h]
                                 + mask[(size_t)(w & 63) * WIN_N * WIN_N + row * WIN_N + col];
                    }
                    Sf[row][col] = s;
                }
            }
        }
        __syncthreads();
        // ---- softmax per row over j<49; probs bf16, padded cols -> 0 ----
        if (tid < NP) {
            int row = tid;
            if (row < WIN_N) {
                float m = -1e30f;
                for (int j = 0; j < WIN_N; ++j) m = fmaxf(m, Sf[row][j]);
                float sum = 0.f;
                for (int j = 0; j < WIN_N; ++j) sum += __expf(Sf[row][j] - m);
                float inv = 1.f / sum;
                for (int j = 0; j < WIN_N; ++j)
                    Ps[row][j] = to_bf16(__expf(Sf[row][j] - m) * inv);
                for (int j = WIN_N; j < NP; ++j) Ps[row][j] = to_bf16(0.f);
            } else {
                for (int j = 0; j < NP; ++j) Ps[row][j] = to_bf16(0.f);
            }
        }
        __syncthreads();
        // ---- O = P @ V (64x32), K = 64 in two steps ----
        {
            int nt = nhalf;
            f32x8 o = {};
            for (int ks = 0; ks < NP; ks += 32) {
                bf16x16 a = load_frag(&Ps[0][0], mt * 16 + lrow, NP, ks + kb);
                bf16x16 b = load_frag(&Vt[0][0], nt * 16 + lrow, NP, ks + kb);
                o = wmma_bf16(a, b, o);
            }
            for (int r = 0; r < 8; ++r) {
                int row = mt * 16 + rbase + r;
                int col = nt * 16 + lrow;
                if (row < WIN_N)
                    attn_out[(size_t)(w * WIN_N + row) * DIMC + h * HD + col] = to_bf16(o[r]);
            }
        }
        __syncthreads();
    }
}

// ---------------------------------------------------------------------------
// Kernel 2: output projection GEMM  (200704 x 512) @ (512 x 512) + bias.
// 128x128 block tile, K-step 32.  A-tile staging uses the CDNA5 async
// global->LDS path (ASYNCcnt) when available; proj_w converted fp32->bf16
// while staging B^T.
// ---------------------------------------------------------------------------
__global__ __launch_bounds__(256) void proj_gemm_kernel(
    const __bf16* __restrict__ A, const float* __restrict__ Bw,
    const float* __restrict__ bias, float* __restrict__ out)
{
    __shared__ __align__(16) __bf16 At[128][32];
    __shared__ __align__(16) __bf16 Bt[128][32];   // B^T: Bt[n][k]

    const int tid   = threadIdx.x;
    const int lane  = tid & 31;
    const int wid   = tid >> 5;
    const int lrow  = lane & 15;
    const int kb    = (lane >> 4) * 8;
    const int rbase = (lane >> 4) * 8;
    const size_t m0 = (size_t)(blockIdx.x >> 2) * 128;
    const int    n0 = (blockIdx.x & 3) * 128;

    f32x8 acc[2][4] = {};

    for (int k0 = 0; k0 < DIMC; k0 += 32) {
        { // stage A tile (bf16, 128x32): 32B / thread
            int row = tid >> 1;
            int cg  = (tid & 1) * 16;
            const __bf16* gsrc = A + (m0 + row) * DIMC + k0 + cg;
#if HAVE_ASYNC_LDS
            __builtin_amdgcn_global_load_async_to_lds_b128(
                (GLOBAL_AS v4i_gcc*)gsrc,       (LDS_AS v4i_gcc*)&At[row][cg],     0, 0);
            __builtin_amdgcn_global_load_async_to_lds_b128(
                (GLOBAL_AS v4i_gcc*)(gsrc + 8), (LDS_AS v4i_gcc*)&At[row][cg + 8], 0, 0);
#else
            const uint4* src = reinterpret_cast<const uint4*>(gsrc);
            uint4 u0 = src[0], u1 = src[1];
            uint4* dst = reinterpret_cast<uint4*>(&At[row][cg]);
            dst[0] = u0; dst[1] = u1;
#endif
        }
        // stage B^T tile from fp32 proj_w, converting to bf16
        for (int i = 0; i < 2; ++i) {
            int e  = tid + i * 256;         // 512 chunks of 8
            int kk = e >> 4;                // 0..31
            int ng = e & 15;                // n-group of 8
            const float4* src = reinterpret_cast<const float4*>(
                Bw + (size_t)(k0 + kk) * DIMC + n0 + ng * 8);
            float4 f0 = src[0], f1 = src[1];
            float vv[8] = {f0.x, f0.y, f0.z, f0.w, f1.x, f1.y, f1.z, f1.w};
            for (int j = 0; j < 8; ++j) Bt[ng * 8 + j][kk] = to_bf16(vv[j]);
        }
        if (k0 + 32 < DIMC) {   // prefetch next A rows (HBM stream)
            __builtin_prefetch(A + (m0 + (tid >> 1)) * DIMC + k0 + 32, 0, 1);
        }
#if HAVE_ASYNC_LDS
        __builtin_amdgcn_s_wait_asynccnt(0);
#endif
        __syncthreads();
        int mrb = (wid & 3) * 32;
        int ncb = (wid >> 2) * 64;
        bf16x16 a0 = load_frag(&At[0][0], mrb + lrow,      32, kb);
        bf16x16 a1 = load_frag(&At[0][0], mrb + 16 + lrow, 32, kb);
        for (int jn = 0; jn < 4; ++jn) {
            bf16x16 b = load_frag(&Bt[0][0], ncb + jn * 16 + lrow, 32, kb);
            acc[0][jn] = wmma_bf16(a0, b, acc[0][jn]);
            acc[1][jn] = wmma_bf16(a1, b, acc[1][jn]);
        }
        __syncthreads();
    }
    int mrb = (wid & 3) * 32;
    int ncb = (wid >> 2) * 64;
    for (int im = 0; im < 2; ++im)
        for (int jn = 0; jn < 4; ++jn)
            for (int r = 0; r < 8; ++r) {
                size_t row = m0 + mrb + im * 16 + rbase + r;
                int    col = n0 + ncb + jn * 16 + lrow;
                out[row * DIMC + col] = acc[im][jn][r] + bias[col];
            }
}

extern "C" void kernel_launch(void* const* d_in, const int* in_sizes, int n_in,
                              void* d_out, int out_size, void* d_ws, size_t ws_size,
                              hipStream_t stream) {
    const float* x          = (const float*)d_in[0];
    const float* mask       = (const float*)d_in[1];
    const float* qkv_w      = (const float*)d_in[2];
    const float* qkv_b      = (const float*)d_in[3];
    const float* proj_w     = (const float*)d_in[4];
    const float* proj_b     = (const float*)d_in[5];
    const float* bias_table = (const float*)d_in[6];

    __bf16* attn_out = (__bf16*)d_ws;   // 200704 * 512 * 2B = 205.5 MB

    swin_qkv_attn_kernel<<<4096, 256, 0, stream>>>(
        x, mask, qkv_w, qkv_b, bias_table, attn_out);

    const int mblocks = 200704 / 128;   // 1568
    proj_gemm_kernel<<<mblocks * 4, 256, 0, stream>>>(
        attn_out, proj_w, proj_b, (float*)d_out);
}
